// GATAutoEncoderDecoder_21406117003703
// MI455X (gfx1250) — matrix operand order repaired and checked by reading, not compiled, not run on hardware
//
#include <hip/hip_runtime.h>

// ---------------- problem constants (from reference) ----------------
constexpr int   kN   = 20000;        // nodes
constexpr int   kE   = 400000;       // edges
constexpr int   kET  = kE + kN;      // edges + self loops
constexpr int   kH   = 4;            // heads
constexpr float kBnEps = 1e-5f;

typedef __attribute__((ext_vector_type(16))) __bf16 v16bf;
typedef __attribute__((ext_vector_type(8)))  __bf16 v8bf;
typedef __attribute__((ext_vector_type(8)))  float  v8f;

__device__ __forceinline__ void atomicMaxF(float* addr, float val) {
  if (val >= 0.0f) atomicMax((int*)addr, __float_as_int(val));
  else             atomicMin((unsigned int*)addr, __float_as_uint(val));
}

// ---------------- elementwise helpers ----------------
__global__ void k_fill(float* __restrict__ p, float v, size_t n) {
  size_t i = (size_t)blockIdx.x * blockDim.x + threadIdx.x;
  if (i < n) p[i] = v;
}

__global__ void k_f32_bf16(const float* __restrict__ s, __bf16* __restrict__ d, size_t n) {
  size_t i = (size_t)blockIdx.x * blockDim.x + threadIdx.x;
  if (i < n) d[i] = (__bf16)s[i];
}

// leaky-relu (slope=0 -> relu) f32 -> bf16
__global__ void k_lrelu_bf16(const float* __restrict__ s, __bf16* __restrict__ d,
                             float slope, size_t n) {
  size_t i = (size_t)blockIdx.x * blockDim.x + threadIdx.x;
  if (i >= n) return;
  float v = s[i];
  d[i] = (__bf16)(v > 0.0f ? v : slope * v);
}

// relu(bn(z)) -> bf16
__global__ void k_bn_relu_bf16(const float* __restrict__ z, __bf16* __restrict__ d,
                               const float* __restrict__ g, const float* __restrict__ be,
                               const float* __restrict__ mean, const float* __restrict__ var,
                               int cmask, size_t n) {
  size_t i = (size_t)blockIdx.x * blockDim.x + threadIdx.x;
  if (i >= n) return;
  int c = (int)(i & (size_t)cmask);
  float t = (z[i] - mean[c]) * rsqrtf(var[c] + kBnEps) * g[c] + be[c];
  d[i] = (__bf16)fmaxf(t, 0.0f);
}

// bn(relu(x)) -> bf16
__global__ void k_relu_bn_bf16(const float* __restrict__ s, __bf16* __restrict__ d,
                               const float* __restrict__ g, const float* __restrict__ be,
                               const float* __restrict__ mean, const float* __restrict__ var,
                               int cmask, size_t n) {
  size_t i = (size_t)blockIdx.x * blockDim.x + threadIdx.x;
  if (i >= n) return;
  int c = (int)(i & (size_t)cmask);
  float t = fmaxf(s[i], 0.0f);
  t = (t - mean[c]) * rsqrtf(var[c] + kBnEps) * g[c] + be[c];
  d[i] = (__bf16)t;
}

// out[n, j] = b[j]   (bias broadcast init before scatter-add)
__global__ void k_rows_bias(float* __restrict__ out, const float* __restrict__ b,
                            int cmask, size_t n) {
  size_t i = (size_t)blockIdx.x * blockDim.x + threadIdx.x;
  if (i < n) out[i] = b[i & (size_t)cmask];
}

// ---------------- graph preprocessing ----------------
__global__ void k_deg_sum(const int* __restrict__ ei, const float* __restrict__ ew,
                          float* __restrict__ deg, float* __restrict__ esum) {
  int e = blockIdx.x * blockDim.x + threadIdx.x;
  if (e >= kE) return;
  int c = ei[kE + e];
  atomicAdd(&deg[c], 1.0f);
  atomicAdd(&esum[c], ew[e]);
}

__global__ void k_edges(const int* __restrict__ ei, const float* __restrict__ ew,
                        int* __restrict__ row, int* __restrict__ col, float* __restrict__ ea) {
  int e = blockIdx.x * blockDim.x + threadIdx.x;
  if (e >= kE) return;
  row[e] = ei[e];
  col[e] = ei[kE + e];
  ea[e]  = ew[e];
}

__global__ void k_loops(int* __restrict__ row, int* __restrict__ col, float* __restrict__ ea,
                        const float* __restrict__ deg, const float* __restrict__ esum) {
  int n = blockIdx.x * blockDim.x + threadIdx.x;
  if (n >= kN) return;
  row[kE + n] = n;
  col[kE + n] = n;
  ea[kE + n]  = esum[n] / fmaxf(deg[n], 1.0f);
}

// ---------------- B-matrix fragment packing ----------------
// Rearranges f32 weight W[K,N] (row-major) into bf16 WMMA-B fragment order:
//   Bp[(((kt*ntiles + nt)*32 + lane)*16) + i] = (bf16) W[kt*32 + (lane>>4)*8 + (i&7) + (i>>3)*16][nt*16 + (lane&15)]
// so a lane's 16-element fragment is one contiguous 32-byte load in the GEMM.
__global__ void k_pack_b(const float* __restrict__ W, __bf16* __restrict__ Bp, int N, int K) {
  int idx = blockIdx.x * blockDim.x + threadIdx.x;
  if (idx >= K * N) return;
  int i      = idx & 15;
  int lane   = (idx >> 4) & 31;
  int t      = idx >> 9;               // kt*ntiles + nt
  int ntiles = N >> 4;
  int nt     = t % ntiles;
  int kt     = t / ntiles;
  int k      = kt * 32 + (lane >> 4) * 8 + (i & 7) + (i >> 3) * 16;
  int col    = nt * 16 + (lane & 15);
  Bp[idx] = (__bf16)W[(size_t)k * N + col];
}

// ---------------- WMMA GEMM: C[M,N] = A[M,K](bf16) @ B[K,N](packed bf16) (+bias) --------
// One wave -> 16x64 of C (4 n-tiles, one A fragment reused across 4 WMMAs per K-step).
// 8 waves/block tile along M. All M,K mult of 16/32; all N mult of 64.
__global__ __launch_bounds__(256) void k_gemm_bf16(
    const __bf16* __restrict__ A, const __bf16* __restrict__ Bp,
    const float* __restrict__ bias, float* __restrict__ C,
    int M, int N, int K) {
  const int wave  = threadIdx.x >> 5;
  const int lane  = threadIdx.x & 31;
  const int mtile = blockIdx.y * 8 + wave;
  if (mtile * 16 >= M) return;                 // wave-uniform guard (EXEC stays all-1s)
  const int ntiles = N >> 4;
  const int nt0    = blockIdx.x * 4;
  const int row0   = mtile * 16;
  const int half   = lane >> 4;
  const int l15    = lane & 15;
  const int arow   = row0 + l15;
  v8f acc0 = {}, acc1 = {}, acc2 = {}, acc3 = {};
  for (int k0 = 0, kt = 0; k0 < K; k0 += 32, ++kt) {
    // A 16x32 bf16: lane<16 holds K[k0..k0+7] and K[k0+16..k0+23]; lane>=16 shifted by 8.
    const int ka = k0 + half * 8;
    v8bf alo = *(const v8bf*)(A + (size_t)arow * K + ka);
    v8bf ahi = *(const v8bf*)(A + (size_t)arow * K + ka + 16);
    v16bf af = __builtin_shufflevector(alo, ahi,
        0, 1, 2, 3, 4, 5, 6, 7, 8, 9, 10, 11, 12, 13, 14, 15);
    // Packed B: 512 bf16 per (kt,nt) tile; lane fragment contiguous.
    const __bf16* bb = Bp + (((size_t)kt * ntiles + nt0) * 32 + lane) * 16;
    v16bf b0 = *(const v16bf*)(bb);
    v16bf b1 = *(const v16bf*)(bb + 512);
    v16bf b2 = *(const v16bf*)(bb + 1024);
    v16bf b3 = *(const v16bf*)(bb + 1536);
    acc0 = __builtin_amdgcn_wmma_f32_16x16x32_bf16(false, af, false, b0, (short)0, acc0, false, false);
    acc1 = __builtin_amdgcn_wmma_f32_16x16x32_bf16(false, af, false, b1, (short)0, acc1, false, false);
    acc2 = __builtin_amdgcn_wmma_f32_16x16x32_bf16(false, af, false, b2, (short)0, acc2, false, false);
    acc3 = __builtin_amdgcn_wmma_f32_16x16x32_bf16(false, af, false, b3, (short)0, acc3, false, false);
  }
  v8f accs[4] = {acc0, acc1, acc2, acc3};
#pragma unroll
  for (int t = 0; t < 4; ++t) {
    const int bcol = (nt0 + t) * 16 + l15;
    const float bv = bias ? bias[bcol] : 0.0f;
#pragma unroll
    for (int v = 0; v < 8; ++v)   // C: VGPR v -> M=row0+half*8+v, N=bcol
      C[(size_t)(row0 + half * 8 + v) * N + bcol] = accs[t][v] + bv;
  }
}

// ---------------- GAT attention ----------------
// a_src[n,h] = sum_c xp[n,h,c]*att_src[h,c];  likewise a_dst.
__global__ void k_att_scores(const float* __restrict__ xp,
                             const float* __restrict__ att_src,
                             const float* __restrict__ att_dst,
                             float* __restrict__ a_src, float* __restrict__ a_dst,
                             int cout) {
  int i = blockIdx.x * blockDim.x + threadIdx.x;
  if (i >= kN * kH) return;
  int n = i >> 2, h = i & 3;
  const float* xr = xp + ((size_t)n * kH + h) * cout;
  const float* as = att_src + h * cout;
  const float* ad = att_dst + h * cout;
  float s = 0.0f, d = 0.0f;
  for (int c = 0; c < cout; ++c) { s += xr[c] * as[c]; d += xr[c] * ad[c]; }
  a_src[i] = s;
  a_dst[i] = d;
}

// wedot[h] = sum_c W_edge[h,c]*att_edge[h,c]  (eattr is scalar per edge)
__global__ void k_wedot(const float* __restrict__ W_edge, const float* __restrict__ att_edge,
                        float* __restrict__ wedot, int cout) {
  int h = threadIdx.x;
  if (h >= kH) return;
  float s = 0.0f;
  for (int c = 0; c < cout; ++c) s += W_edge[h * cout + c] * att_edge[h * cout + c];
  wedot[h] = s;
}

__global__ void k_alpha_max(const float* __restrict__ a_src, const float* __restrict__ a_dst,
                            const float* __restrict__ ea, const float* __restrict__ wedot,
                            const int* __restrict__ row, const int* __restrict__ col,
                            float* __restrict__ alpha, float* __restrict__ mx) {
  int i = blockIdx.x * blockDim.x + threadIdx.x;
  if (i >= kET * kH) return;
  int e = i >> 2, h = i & 3;
  float v = a_src[row[e] * kH + h] + a_dst[col[e] * kH + h] + ea[e] * wedot[h];
  v = v > 0.0f ? v : 0.2f * v;                 // attention LeakyReLU
  alpha[i] = v;
  atomicMaxF(&mx[col[e] * kH + h], v);
}

__global__ void k_exp_sum(float* __restrict__ alpha, const float* __restrict__ mx,
                          float* __restrict__ denom, const int* __restrict__ col) {
  int i = blockIdx.x * blockDim.x + threadIdx.x;
  if (i >= kET * kH) return;
  int e = i >> 2, h = i & 3;
  float ex = __expf(alpha[i] - mx[col[e] * kH + h]);
  alpha[i] = ex;
  atomicAdd(&denom[col[e] * kH + h], ex);
}

__global__ void k_alpha_norm(float* __restrict__ alpha, const float* __restrict__ denom,
                             const int* __restrict__ col) {
  int i = blockIdx.x * blockDim.x + threadIdx.x;
  if (i >= kET * kH) return;
  int e = i >> 2, h = i & 3;
  alpha[i] /= denom[col[e] * kH + h];
}

// out[col[e], j] += xp[row[e], j] * alpha[e, j>>cshift]; HC = 1<<hcshift.
// For HC=256 a whole 256-thread block shares one edge -> coalesced gather + atomic row.
__global__ void k_scatter(const float* __restrict__ xp, const float* __restrict__ alpha,
                          const int* __restrict__ row, const int* __restrict__ col,
                          float* __restrict__ out, int hcshift, int cshift) {
  size_t i = (size_t)blockIdx.x * blockDim.x + threadIdx.x;
  size_t tot = (size_t)kET << hcshift;
  if (i >= tot) return;
  int e = (int)(i >> hcshift);
  int j = (int)(i & (((size_t)1 << hcshift) - 1));
  int h = j >> cshift;
  float m = xp[((size_t)row[e] << hcshift) + j] * alpha[(size_t)e * kH + h];
  atomicAdd(&out[((size_t)col[e] << hcshift) + j], m);
}

// ---------------- launcher ----------------
extern "C" void kernel_launch(void* const* d_in, const int* in_sizes, int n_in,
                              void* d_out, int out_size, void* d_ws, size_t ws_size,
                              hipStream_t stream) {
  (void)in_sizes; (void)n_in; (void)out_size; (void)ws_size;
  const float* x  = (const float*)d_in[0];
  const int*   ei = (const int*)  d_in[1];
  const float* ew = (const float*)d_in[2];
  const float *gW[4], *gAs[4], *gAd[4], *gWe[4], *gAe[4], *gB[4];
  for (int i = 0; i < 4; ++i) {
    gW[i]  = (const float*)d_in[3 + 6 * i + 0];
    gAs[i] = (const float*)d_in[3 + 6 * i + 1];
    gAd[i] = (const float*)d_in[3 + 6 * i + 2];
    gWe[i] = (const float*)d_in[3 + 6 * i + 3];
    gAe[i] = (const float*)d_in[3 + 6 * i + 4];
    gB[i]  = (const float*)d_in[3 + 6 * i + 5];
  }
  const float* eW  = (const float*)d_in[27];
  const float* eB  = (const float*)d_in[28];
  const float* eG  = (const float*)d_in[29];
  const float* eBe = (const float*)d_in[30];
  const float* eM  = (const float*)d_in[31];
  const float* eV  = (const float*)d_in[32];
  const float* dW1 = (const float*)d_in[33];
  const float* dB1 = (const float*)d_in[34];
  const float* dG  = (const float*)d_in[35];
  const float* dBe = (const float*)d_in[36];
  const float* dM  = (const float*)d_in[37];
  const float* dV  = (const float*)d_in[38];
  const float* dW2 = (const float*)d_in[39];
  const float* dB2 = (const float*)d_in[40];
  const float* dW3 = (const float*)d_in[41];
  const float* dB3 = (const float*)d_in[42];

  // per-layer dims
  const int cin_[4]  = {128, 128, 64, 256};
  const int cout_[4] = {32, 16, 64, 64};
  const int hc_[4]   = {128, 64, 256, 256};
  const int csh_[4]  = {5, 4, 6, 6};     // log2(cout)
  const int hcsh_[4] = {7, 6, 8, 8};     // log2(H*cout)

  // workspace bump allocator (~64 MB total)
  char* wp = (char*)d_ws;
  auto bump = [&](size_t bytes) -> char* {
    char* r = wp;
    wp += (bytes + 255) & ~(size_t)255;
    return r;
  };
  int*    rowA  = (int*)   bump(sizeof(int) * kET);
  int*    colA  = (int*)   bump(sizeof(int) * kET);
  float*  eattr = (float*) bump(sizeof(float) * kET);
  float*  deg   = (float*) bump(sizeof(float) * kN);
  float*  esum  = (float*) bump(sizeof(float) * kN);
  float*  wedot = (float*) bump(sizeof(float) * kH);
  float*  a_src = (float*) bump(sizeof(float) * kN * kH);
  float*  a_dst = (float*) bump(sizeof(float) * kN * kH);
  float*  mx    = (float*) bump(sizeof(float) * kN * kH);
  float*  denom = (float*) bump(sizeof(float) * kN * kH);
  float*  alpha = (float*) bump(sizeof(float) * (size_t)kET * kH);
  float*  xp    = (float*) bump(sizeof(float) * (size_t)kN * 256);
  float*  hout  = (float*) bump(sizeof(float) * (size_t)kN * 256);
  __bf16* habf  = (__bf16*)bump(sizeof(__bf16) * (size_t)kN * 256);
  __bf16* wbf[4];
  for (int i = 0; i < 4; ++i)
    wbf[i] = (__bf16*)bump(sizeof(__bf16) * (size_t)cin_[i] * hc_[i]);
  __bf16* webf = (__bf16*)bump(sizeof(__bf16) * 256 * 64);
  __bf16* w1bf = (__bf16*)bump(sizeof(__bf16) * 64 * 128);
  __bf16* w2bf = (__bf16*)bump(sizeof(__bf16) * 128 * 64);
  __bf16* w3bf = (__bf16*)bump(sizeof(__bf16) * 64 * 128);

  auto g1 = [](size_t n) { return (unsigned)((n + 255) / 256); };
  auto gemm = [&](const __bf16* A, const __bf16* Bp, const float* bias, float* C,
                  int M, int Nc, int K) {
    dim3 grid((unsigned)(Nc / 64), (unsigned)((M / 16 + 7) / 8));
    k_gemm_bf16<<<grid, 256, 0, stream>>>(A, Bp, bias, C, M, Nc, K);
  };

  // weights -> packed bf16 fragments (small; recomputed per call for determinism)
  for (int i = 0; i < 4; ++i) {
    size_t n = (size_t)cin_[i] * hc_[i];
    k_pack_b<<<g1(n), 256, 0, stream>>>(gW[i], wbf[i], hc_[i], cin_[i]);
  }
  k_pack_b<<<g1(256 * 64), 256, 0, stream>>>(eW,  webf, 64, 256);
  k_pack_b<<<g1(64 * 128), 256, 0, stream>>>(dW1, w1bf, 128, 64);
  k_pack_b<<<g1(128 * 64), 256, 0, stream>>>(dW2, w2bf, 64, 128);
  k_pack_b<<<g1(64 * 128), 256, 0, stream>>>(dW3, w3bf, 128, 64);

  // graph prep: degree, mean incoming edge weight for self-loops, augmented edge list
  k_fill<<<g1(kN), 256, 0, stream>>>(deg,  0.0f, kN);
  k_fill<<<g1(kN), 256, 0, stream>>>(esum, 0.0f, kN);
  k_deg_sum<<<g1(kE), 256, 0, stream>>>(ei, ew, deg, esum);
  k_edges<<<g1(kE), 256, 0, stream>>>(ei, ew, rowA, colA, eattr);
  k_loops<<<g1(kN), 256, 0, stream>>>(rowA, colA, eattr, deg, esum);

  // x -> bf16 activations
  k_f32_bf16<<<g1((size_t)kN * 128), 256, 0, stream>>>(x, habf, (size_t)kN * 128);

  // ---- 4 GAT layers ----
  for (int i = 0; i < 4; ++i) {
    const int HC = hc_[i];
    gemm(habf, wbf[i], nullptr, xp, kN, HC, cin_[i]);                       // xp = h @ W
    k_att_scores<<<g1(kN * kH), 256, 0, stream>>>(xp, gAs[i], gAd[i], a_src, a_dst, cout_[i]);
    k_wedot<<<1, 32, 0, stream>>>(gWe[i], gAe[i], wedot, cout_[i]);
    k_fill<<<g1(kN * kH), 256, 0, stream>>>(mx,   -1e30f, kN * kH);
    k_fill<<<g1(kN * kH), 256, 0, stream>>>(denom, 0.0f,  kN * kH);
    k_alpha_max<<<g1((size_t)kET * kH), 256, 0, stream>>>(a_src, a_dst, eattr, wedot,
                                                          rowA, colA, alpha, mx);
    k_exp_sum<<<g1((size_t)kET * kH), 256, 0, stream>>>(alpha, mx, denom, colA);
    k_alpha_norm<<<g1((size_t)kET * kH), 256, 0, stream>>>(alpha, denom, colA);
    k_rows_bias<<<g1((size_t)kN * HC), 256, 0, stream>>>(hout, gB[i], HC - 1, (size_t)kN * HC);
    k_scatter<<<g1((size_t)kET << hcsh_[i]), 256, 0, stream>>>(xp, alpha, rowA, colA,
                                                               hout, hcsh_[i], csh_[i]);
    k_lrelu_bf16<<<g1((size_t)kN * HC), 256, 0, stream>>>(hout, habf, 0.01f, (size_t)kN * HC);
  }

  // ---- embedding + decoder ----
  float* recon = (float*)d_out;                        // (N,128)
  float* z     = recon + (size_t)kN * 128;             // (N,64), second tuple output
  gemm(habf, webf, eB, z, kN, 64, 256);                                     // z = h@W_emb+b
  k_bn_relu_bf16<<<g1((size_t)kN * 64), 256, 0, stream>>>(z, habf, eG, eBe, eM, eV,
                                                          63, (size_t)kN * 64);
  gemm(habf, w1bf, dB1, hout, kN, 128, 64);                                 // h@W1+b1
  k_relu_bn_bf16<<<g1((size_t)kN * 128), 256, 0, stream>>>(hout, habf, dG, dBe, dM, dV,
                                                           127, (size_t)kN * 128);
  gemm(habf, w2bf, dB2, hout, kN, 64, 128);                                 // h1@W2+b2
  k_lrelu_bf16<<<g1((size_t)kN * 64), 256, 0, stream>>>(hout, habf, 0.0f, (size_t)kN * 64);
  gemm(habf, w3bf, dB3, recon, kN, 128, 64);                                // recon = h2@W3+b3
}
